// Attention_27298812133822
// MI455X (gfx1250) — matrix operand order repaired and checked by reading, not compiled, not run on hardware
//
#include <hip/hip_runtime.h>
#include <hip/hip_bf16.h>

#define B_  4
#define S_  2048
#define DM  768     // d_model
#define NHD 12      // n_heads
#define DH  64      // d_head

typedef __bf16 bf16;
typedef __bf16 v16bf __attribute__((ext_vector_type(16)));
typedef __bf16 v8bf  __attribute__((ext_vector_type(8)));
typedef float  v8f   __attribute__((ext_vector_type(8)));
typedef float  f32x8 __attribute__((ext_vector_type(8)));

static __device__ __forceinline__ v8f wmma_bf16(v16bf a, v16bf b, v8f c) {
    // (neg_a, A, neg_b, B, c_mod, C, reuse_a, reuse_b)
    return __builtin_amdgcn_wmma_f32_16x16x32_bf16(false, a, false, b, (short)0, c, false, false);
}

// ---------------------------------------------------------------------------
// Kernel 0: one-time weight prep (fp32 -> bf16, transposed for contiguous
// B-fragment loads).
//   Wt  [qkv][n][h][m]  <- W_{qkv}[n][m][h]
//   WoT [m][n*h]        <- W_O[n][h][m]
// ---------------------------------------------------------------------------
__global__ __launch_bounds__(256)
void prep_weights_kernel(const float* __restrict__ Wq, const float* __restrict__ Wk,
                         const float* __restrict__ Wv, const float* __restrict__ Wo,
                         bf16* __restrict__ Wt, bf16* __restrict__ WoT)
{
    const int NW = NHD * DH * DM;                       // 589824
    int id = blockIdx.x * blockDim.x + threadIdx.x;     // total = 4*NW
    if (id < 3 * NW) {
        int qkv = id / NW;
        int r   = id % NW;            // dst layout [n][h][m], m fastest (coalesced write)
        int m = r % DM;
        int h = (r / DM) % DH;
        int n = r / (DM * DH);
        const float* W = (qkv == 0) ? Wq : (qkv == 1) ? Wk : Wv;
        Wt[id] = (bf16)W[(n * DM + m) * DH + h];
    } else {
        int e = id - 3 * NW;          // dst layout [m][nh], nh fastest
        int k = e % (NHD * DH);
        int m = e / (NHD * DH);
        WoT[e] = (bf16)Wo[k * DM + m];
    }
}

// ---------------------------------------------------------------------------
// Kernel 1: fused QKV projection. Each wave computes a 16x64 strip (one head
// of one of Q/K/V): 4 WMMA C-tiles sharing one A fragment per k-step.
// Q,K stored [b][n][s][h]; V stored transposed [b][n][h][s] for the PV GEMM.
// ---------------------------------------------------------------------------
__global__ __launch_bounds__(256)
void qkv_proj_kernel(const float* __restrict__ x,
                     const bf16*  __restrict__ Wt,    // [qkv][n][h][m]
                     const float* __restrict__ bq,
                     const float* __restrict__ bk,
                     const float* __restrict__ bv,
                     bf16* __restrict__ Qws, bf16* __restrict__ Kws, bf16* __restrict__ Vws)
{
    const int lane = threadIdx.x & 31;
    const int wave = threadIdx.x >> 5;
    const int half = lane >> 4;
    const int l16  = lane & 15;

    const int rowTile  = blockIdx.y * 8 + wave;   // 0..511 over B*S rows
    const int colGroup = blockIdx.x;              // 0..35 == qkv*NHD + head
    const int qkv  = colGroup / NHD;              // uniform per wave
    const int head = colGroup % NHD;              // uniform per wave
    const int row0 = rowTile * 16;
    const int rowA = row0 + l16;

    const float* bias = (qkv == 0) ? bq : (qkv == 1) ? bk : bv;
    bf16*        dst  = (qkv == 0) ? Qws : (qkv == 1) ? Kws : Vws;
    const bf16*  Wb   = Wt + (size_t)(qkv * NHD + head) * DH * DM;   // [h][m]

    v8f acc[4] = {};
    for (int kt = 0; kt < DM; kt += 32) {
        // A fragment (16x32 bf16 layout): two contiguous runs of 8 fp32, convert.
        const f32x8* xp = (const f32x8*)(x + (size_t)rowA * DM + kt + half * 16);
        f32x8 a0 = xp[0];   // k = kt + half*16 + 0..7
        f32x8 a1 = xp[1];   // k = kt + half*16 + 8..15
        v16bf av;
        #pragma unroll
        for (int i = 0; i < 8; ++i) { av[i] = (bf16)a0[i]; av[i + 8] = (bf16)a1[i]; }
        // 4 B fragments, each a single contiguous v16bf load (k = kt+half*16+i).
        #pragma unroll
        for (int sub = 0; sub < 4; ++sub) {
            const v16bf* wp =
                (const v16bf*)(Wb + (size_t)(sub * 16 + l16) * DM + kt + half * 16);
            acc[sub] = wmma_bf16(av, wp[0], acc[sub]);
        }
    }
    #pragma unroll
    for (int sub = 0; sub < 4; ++sub) {
        const int h = sub * 16 + l16;
        const float bval = bias[head * DH + h];
        #pragma unroll
        for (int r = 0; r < 8; ++r) {
            int row = row0 + r + 8 * half;      // global b*S + s
            int b = row / S_, s = row % S_;
            float v = acc[sub][r] + bval;
            if (qkv == 2) {
                // V transposed: [b][n][h][s]
                dst[((size_t)(b * NHD + head) * DH + h) * S_ + s] = (bf16)v;
            } else {
                // Q/K: [b][n][s][h]
                dst[((size_t)(b * NHD + head) * S_ + s) * DH + h] = (bf16)v;
            }
        }
    }
}

// ---------------------------------------------------------------------------
// Kernel 2: flash attention. One 16-query tile per wave, streaming 32-key
// blocks with online softmax. P transposed to A-layout via per-wave LDS tile.
// V is pre-transposed [h][s] so PV B-fragments are contiguous v16bf loads.
// ---------------------------------------------------------------------------
#define PSTRIDE 34   // padded LDS row stride (elements) to avoid bank conflicts

__global__ __launch_bounds__(256)
void flash_attn_kernel(const bf16* __restrict__ Qws,    // [b][n][s][h]
                       const bf16* __restrict__ Kws,    // [b][n][s][h]
                       const bf16* __restrict__ Vws,    // [b][n][h][s]
                       bf16* __restrict__ Zws)          // [b*s][n*h]
{
    __shared__ bf16 Plds[8][16 * PSTRIDE];

    const int lane = threadIdx.x & 31;
    const int wave = threadIdx.x >> 5;
    const int half = lane >> 4;
    const int l16  = lane & 15;

    const int bn    = blockIdx.y;                 // 0..47 == b*NHD + n
    const int qTile = blockIdx.x * 8 + wave;      // 0..127
    const int q0    = qTile * 16;

    const bf16* Qb = Qws + (size_t)bn * S_ * DH;
    const bf16* Kb = Kws + (size_t)bn * S_ * DH;
    const bf16* Vb = Vws + (size_t)bn * DH * S_;

    // Q tile in A-layout: two K=32 chunks over the head dim, held for the loop.
    v16bf qa[2];
    {
        const int qrow = q0 + l16;
        #pragma unroll
        for (int c2 = 0; c2 < 2; ++c2) {
            const v8bf* qp = (const v8bf*)(Qb + (size_t)qrow * DH + c2 * 32 + half * 16);
            v8bf lo = qp[0];
            v8bf hi = qp[1];   // +8 elements
            #pragma unroll
            for (int i = 0; i < 8; ++i) { qa[c2][i] = lo[i]; qa[c2][i + 8] = hi[i]; }
        }
    }

    v8f o[4] = {};
    float mrow[8], lrow[8];
    #pragma unroll
    for (int r = 0; r < 8; ++r) { mrow[r] = -__builtin_inff(); lrow[r] = 0.f; }

    const float scale = 0.125f;   // 1/sqrt(64)

    for (int kb = 0; kb < q0 + 16; kb += 32) {
        float st[2][8];
        #pragma unroll
        for (int t = 0; t < 2; ++t) {
            const int key = kb + t * 16 + l16;    // B-layout: lane = key column
            const v16bf* kp = (const v16bf*)(Kb + (size_t)key * DH + half * 16);
            v16bf kv0 = kp[0];                    // h = half*16 + i
            v16bf kv1 = kp[2];                    // h = 32 + half*16 + i
            v8f sc = {};
            sc = wmma_bf16(qa[0], kv0, sc);
            sc = wmma_bf16(qa[1], kv1, sc);
            #pragma unroll
            for (int r = 0; r < 8; ++r) {
                int q = q0 + r + 8 * half;
                float v = sc[r] * scale;
                st[t][r] = (key > q) ? -__builtin_inff() : v;
            }
        }
        // Online softmax update (row stats reduced over the 16 lanes of each half).
        #pragma unroll
        for (int r = 0; r < 8; ++r) {
            float m = fmaxf(st[0][r], st[1][r]);
            #pragma unroll
            for (int off = 8; off >= 1; off >>= 1)
                m = fmaxf(m, __shfl_xor(m, off, 32));
            float mn = fmaxf(mrow[r], m);
            float alpha = __expf(mrow[r] - mn);          // 0 on first block
            float p0 = __expf(st[0][r] - mn);
            float p1 = __expf(st[1][r] - mn);
            float rs = p0 + p1;
            #pragma unroll
            for (int off = 8; off >= 1; off >>= 1)
                rs += __shfl_xor(rs, off, 32);
            lrow[r] = lrow[r] * alpha + rs;
            mrow[r] = mn;
            #pragma unroll
            for (int ht = 0; ht < 4; ++ht) o[ht][r] *= alpha;
            // Stage P (row-major 16x32, padded) for the A-layout re-read.
            int prow = r + 8 * half;
            Plds[wave][prow * PSTRIDE + l16]      = (bf16)p0;
            Plds[wave][prow * PSTRIDE + 16 + l16] = (bf16)p1;
        }
        // Cross-lane RAW through LDS within this wave: wait for the DS writes.
        asm volatile("s_wait_dscnt 0" ::: "memory");
        // Read P back in the 16-bit A-matrix layout (K = 32 keys of this block).
        v16bf pa;
        #pragma unroll
        for (int i = 0; i < 16; ++i) {
            int kcol = half * 16 + i;
            pa[i] = Plds[wave][l16 * PSTRIDE + kcol];
        }
        // P x V: four 16-wide head tiles; B fragments contiguous from V^T.
        #pragma unroll
        for (int ht = 0; ht < 4; ++ht) {
            const v16bf* vp =
                (const v16bf*)(Vb + (size_t)(ht * 16 + l16) * S_ + kb + half * 16);
            o[ht] = wmma_bf16(pa, vp[0], o[ht]);
        }
    }

    // Normalize and store Z as bf16 [b*S + q][n*DH + h] for the output GEMM.
    const int b = bn / NHD, n = bn % NHD;
    #pragma unroll
    for (int r = 0; r < 8; ++r) {
        float inv = 1.0f / lrow[r];
        int q = q0 + r + 8 * half;
        size_t base = (size_t)(b * S_ + q) * (NHD * DH) + n * DH;
        #pragma unroll
        for (int ht = 0; ht < 4; ++ht)
            Zws[base + ht * 16 + l16] = (bf16)(o[ht][r] * inv);
    }
}

// ---------------------------------------------------------------------------
// Kernel 3: output projection. Each wave computes a 16x64 strip.
//   out[8192x768] = Z(bf16) @ WoT^T + b_O   with WoT[m][nh] bf16.
// ---------------------------------------------------------------------------
__global__ __launch_bounds__(256)
void out_proj_kernel(const bf16* __restrict__ Z,      // [b*s][nh]
                     const bf16* __restrict__ WoT,    // [m][nh]
                     const float* __restrict__ bo,
                     float* __restrict__ out)
{
    const int lane = threadIdx.x & 31;
    const int wave = threadIdx.x >> 5;
    const int half = lane >> 4;
    const int l16  = lane & 15;

    const int rowTile  = blockIdx.y * 8 + wave;   // 0..511
    const int colGroup = blockIdx.x;              // 0..11 (64 cols each)
    const int row0 = rowTile * 16;
    const int rowA = row0 + l16;

    v8f acc[4] = {};
    for (int kt = 0; kt < NHD * DH; kt += 32) {
        const v8bf* zp = (const v8bf*)(Z + (size_t)rowA * DM + kt + half * 16);
        v8bf lo = zp[0], hi = zp[1];
        v16bf av;
        #pragma unroll
        for (int i = 0; i < 8; ++i) { av[i] = lo[i]; av[i + 8] = hi[i]; }
        #pragma unroll
        for (int sub = 0; sub < 4; ++sub) {
            const int m = colGroup * 64 + sub * 16 + l16;
            const v16bf* wp = (const v16bf*)(WoT + (size_t)m * DM + kt + half * 16);
            acc[sub] = wmma_bf16(av, wp[0], acc[sub]);
        }
    }
    #pragma unroll
    for (int sub = 0; sub < 4; ++sub) {
        const int col = colGroup * 64 + sub * 16 + l16;
        const float bval = bo[col];
        #pragma unroll
        for (int r = 0; r < 8; ++r) {
            int row = row0 + r + 8 * half;
            out[(size_t)row * DM + col] = acc[sub][r] + bval;
        }
    }
}

// ---------------------------------------------------------------------------
extern "C" void kernel_launch(void* const* d_in, const int* in_sizes, int n_in,
                              void* d_out, int out_size, void* d_ws, size_t ws_size,
                              hipStream_t stream)
{
    const float* x  = (const float*)d_in[0];
    const float* Wq = (const float*)d_in[1];
    const float* Wk = (const float*)d_in[2];
    const float* Wv = (const float*)d_in[3];
    const float* Wo = (const float*)d_in[4];
    const float* bq = (const float*)d_in[5];
    const float* bk = (const float*)d_in[6];
    const float* bv = (const float*)d_in[7];
    const float* bo = (const float*)d_in[8];
    float* out = (float*)d_out;

    const size_t elems = (size_t)B_ * NHD * S_ * DH;   // 6.29M per tensor
    const size_t nw    = (size_t)NHD * DH * DM;        // 589824
    bf16* Qws = (bf16*)d_ws;
    bf16* Kws = Qws + elems;
    bf16* Vws = Kws + elems;                           // stored transposed [b][n][h][s]
    bf16* Zws = Vws + elems;
    bf16* Wt  = Zws + elems;                           // [3][n][h][m]
    bf16* WoT = Wt + 3 * nw;                           // [m][nh]   (~55 MB total)

    // Weight prep: 4*589824 elements, one thread each.
    prep_weights_kernel<<<dim3((unsigned)(4 * nw / 256)), 256, 0, stream>>>(
        Wq, Wk, Wv, Wo, Wt, WoT);
    // QKV: 36 column groups (3*12 heads), 512 row tiles / 8 waves per block.
    qkv_proj_kernel<<<dim3(36, 64), 256, 0, stream>>>(x, Wt, bq, bk, bv,
                                                      Qws, Kws, Vws);
    // Attention: 128 q-tiles / 8 waves, 48 (b,n) pairs.
    flash_attn_kernel<<<dim3(16, 48), 256, 0, stream>>>(Qws, Kws, Vws, Zws);
    // Output projection: 12 column groups, 512 row tiles / 8 waves.
    out_proj_kernel<<<dim3(12, 64), 256, 0, stream>>>(Zws, WoT, bo, out);
}
